// DCTEnhancedUNet_89008902242706
// MI455X (gfx1250) — matrix operand order repaired and checked by reading, not compiled, not run on hardware
//
#include <hip/hip_runtime.h>
#include <cmath>

#define DEV __device__ __forceinline__

typedef _Float16 v16h __attribute__((ext_vector_type(16)));
typedef float    v8f  __attribute__((ext_vector_type(8)));

union F16Frag { v16h v; _Float16 h[16]; };
union F32Acc  { v8f  v; float    f[8];  };

// ---------------------------------------------------------------------------
// CDNA5-specific helpers (guarded so the host pass never sees the builtins)
// ---------------------------------------------------------------------------
DEV v8f wmma_f16(v16h a, v16h b, v8f c) {
#if defined(__gfx1250__)
  // D = A(16x32 f16) * B(32x16 f16) + C(16x16 f32)
  return __builtin_amdgcn_wmma_f32_16x16x32_f16(false, a, false, b, (short)0, c,
                                                false, false);
#else
  (void)a; (void)b;
  return c;
#endif
}

DEV void wait_tensor0() {
#if defined(__gfx1250__) && __has_builtin(__builtin_amdgcn_s_wait_tensorcnt)
  __builtin_amdgcn_s_wait_tensorcnt(0);
#endif
}

DEV void cluster_sync_if_clustered() {
#if defined(__gfx1250__) && __has_builtin(__builtin_amdgcn_cluster_id_x) && \
    __has_builtin(__builtin_amdgcn_s_cluster_barrier)
  if (__builtin_amdgcn_cluster_id_x() != 0) __builtin_amdgcn_s_cluster_barrier();
#endif
}

// TDM prefetch of a [rows x cols] f32 tile into LDS (D# groups 0/1, 2-D tensor).
// clang-22 (ROCm 7.2) builtin takes 5 args; clang-23 takes 6.
DEV void tdm_prefetch(const float* gptr, void* ldsptr, int rows, int cols,
                      int stride) {
#if defined(__gfx1250__) && __has_builtin(__builtin_amdgcn_tensor_load_to_lds)
  typedef unsigned int u32x4 __attribute__((ext_vector_type(4)));
  typedef int          i32x4 __attribute__((ext_vector_type(4)));
  typedef int          i32x8 __attribute__((ext_vector_type(8)));
  unsigned long long ga  = (unsigned long long)(uintptr_t)gptr;
  unsigned int       lds = (unsigned int)(uintptr_t)ldsptr;
  u32x4 g0;
  g0[0] = 1u;                                      // count=1, valid user D#
  g0[1] = lds;                                     // lds_addr
  g0[2] = (unsigned int)(ga & 0xFFFFFFFFull);      // global_addr[31:0]
  g0[3] = (unsigned int)((ga >> 32) & 0x1FFFFFFull) | (2u << 30); // [56:32], type=2
  unsigned long long ucols = (unsigned long long)(unsigned)cols;
  unsigned long long urows = (unsigned long long)(unsigned)rows;
  unsigned long long ustrd = (unsigned long long)(unsigned)stride;
  unsigned long long q0 = (2ull << 16)                  // data_size = 4B
                        | ((ucols & 0xFFFFull) << 48);  // tensor_dim0[15:0]
  unsigned long long q1 = (ucols >> 16)                 // tensor_dim0[31:16]
                        | (urows << 16)                 // tensor_dim1
                        | ((ucols & 0xFFFFull) << 48);  // tile_dim0
  unsigned long long q2 = (urows & 0xFFFFull)           // tile_dim1
                        | (ustrd << 32);                // tensor_dim0_stride[31:0]
  unsigned long long q3 = (ustrd >> 32)                 // stride[47:32]
                        | (ustrd << 16);                // tensor_dim1_stride
  i32x8 g1;
  g1[0] = (int)(q0 & 0xFFFFFFFFull); g1[1] = (int)(q0 >> 32);
  g1[2] = (int)(q1 & 0xFFFFFFFFull); g1[3] = (int)(q1 >> 32);
  g1[4] = (int)(q2 & 0xFFFFFFFFull); g1[5] = (int)(q2 >> 32);
  g1[6] = (int)(q3 & 0xFFFFFFFFull); g1[7] = (int)(q3 >> 32);
  i32x4 g2; g2[0] = 0; g2[1] = 0; g2[2] = 0; g2[3] = 0;
  i32x4 g3; g3[0] = 0; g3[1] = 0; g3[2] = 0; g3[3] = 0;
#if __clang_major__ >= 23
  i32x8 g4;
  g4[0]=0; g4[1]=0; g4[2]=0; g4[3]=0; g4[4]=0; g4[5]=0; g4[6]=0; g4[7]=0;
  __builtin_amdgcn_tensor_load_to_lds(g0, g1, g2, g3, g4, 0);
#else
  __builtin_amdgcn_tensor_load_to_lds(g0, g1, g2, g3, 0);
#endif
#else
  (void)gptr; (void)ldsptr; (void)rows; (void)cols; (void)stride;
#endif
}

DEV float postproc(float v, const float* bias, int m, int act) {
  v += bias ? bias[m] : 0.f;
  if (act == 1)      v = fmaxf(v, 0.f);
  else if (act == 2) v = (v > 20.f) ? v : log1pf(__expf(v));
  return v;
}

// Build an A fragment (16x32, contiguous k order matching b_frag_vec) from 2 contiguous
// 8-float runs: lanes 0-15 hold K [0..15], lanes 16-31 [16..31].
DEV F16Frag a_frag_vec(const float* row, int k0, int aoff) {
  const float4* p = (const float4*)(row + k0 + aoff);
  float4 a0 = p[0], a1 = p[1];
  const float4* q = (const float4*)(row + k0 + aoff + 8);
  float4 a2 = q[0], a3 = q[1];
  F16Frag f;
  f.h[0]=(_Float16)a0.x;  f.h[1]=(_Float16)a0.y;  f.h[2]=(_Float16)a0.z;  f.h[3]=(_Float16)a0.w;
  f.h[4]=(_Float16)a1.x;  f.h[5]=(_Float16)a1.y;  f.h[6]=(_Float16)a1.z;  f.h[7]=(_Float16)a1.w;
  f.h[8]=(_Float16)a2.x;  f.h[9]=(_Float16)a2.y;  f.h[10]=(_Float16)a2.z; f.h[11]=(_Float16)a2.w;
  f.h[12]=(_Float16)a3.x; f.h[13]=(_Float16)a3.y; f.h[14]=(_Float16)a3.z; f.h[15]=(_Float16)a3.w;
  return f;
}

// B fragment (32x16): lane = column, lanes 0-15 K 0..15, lanes 16-31 K 16..31:
// 16 contiguous floats.
DEV F16Frag b_frag_vec(const float* row, int k0, int boff) {
  const float4* p = (const float4*)(row + k0 + boff);
  float4 b0 = p[0], b1 = p[1], b2 = p[2], b3 = p[3];
  F16Frag f;
  f.h[0]=(_Float16)b0.x;  f.h[1]=(_Float16)b0.y;  f.h[2]=(_Float16)b0.z;  f.h[3]=(_Float16)b0.w;
  f.h[4]=(_Float16)b1.x;  f.h[5]=(_Float16)b1.y;  f.h[6]=(_Float16)b1.z;  f.h[7]=(_Float16)b1.w;
  f.h[8]=(_Float16)b2.x;  f.h[9]=(_Float16)b2.y;  f.h[10]=(_Float16)b2.z; f.h[11]=(_Float16)b2.w;
  f.h[12]=(_Float16)b3.x; f.h[13]=(_Float16)b3.y; f.h[14]=(_Float16)b3.z; f.h[15]=(_Float16)b3.w;
  return f;
}

// ---------------------------------------------------------------------------
// WMMA GEMM:  C[n*ldc+m] = sum_k A[m*lda+k] * B[n*ldb+k]  (+bias, activation)
// Requires K % 32 == 0 (true for all call sites: 512, 1024, 32).
// Each wave computes a 16(M) x 32(N) tile: A fragment reused across 2 WMMAs.
// Out-of-range rows are clamped; their (unstored) D rows/cols absorb garbage.
// act: 0 = none, 1 = relu, 2 = softplus(x + bias)
// ---------------------------------------------------------------------------
__global__ void k_gemm_wmma(const float* __restrict__ A,
                            const float* __restrict__ B,
                            const float* __restrict__ bias,
                            float* __restrict__ C,
                            int M, int N, int K, int lda, int ldb, int ldc,
                            int act) {
  int lane = threadIdx.x & 31;
  int wave = blockIdx.x * (blockDim.x >> 5) + (threadIdx.x >> 5);
  int mt = (M + 15) >> 4, nt = (N + 31) >> 5;
  if (wave >= mt * nt) return;
  int m0 = (wave % mt) << 4;
  int n0 = (wave / mt) << 5;
  int lm = lane & 15;
  int mi  = min(m0 + lm, M - 1);
  int ni0 = min(n0 + lm, N - 1);
  int ni1 = min(n0 + 16 + lm, N - 1);
  int aoff = (lane < 16) ? 0 : 16;
  int boff = (lane < 16) ? 0 : 16;

  __shared__ float lds_stage[16 * 32];
  if (threadIdx.x == 0) tdm_prefetch(B + (size_t)n0 * ldb, lds_stage, 16, 32, ldb);
  cluster_sync_if_clustered();

  const float* Arow  = A + (size_t)mi * lda;
  const float* Brow0 = B + (size_t)ni0 * ldb;
  const float* Brow1 = B + (size_t)ni1 * ldb;

  F32Acc acc0, acc1;
#pragma unroll
  for (int r = 0; r < 8; ++r) { acc0.f[r] = 0.f; acc1.f[r] = 0.f; }

#pragma unroll 2
  for (int k0 = 0; k0 < K; k0 += 32) {
    if (k0 + 64 < K) __builtin_prefetch(Arow + k0 + 64, 0, 1);
    F16Frag af  = a_frag_vec(Arow, k0, aoff);
    F16Frag bf0 = b_frag_vec(Brow0, k0, boff);
    acc0.v = wmma_f16(af.v, bf0.v, acc0.v);
    F16Frag bf1 = b_frag_vec(Brow1, k0, boff);
    acc1.v = wmma_f16(af.v, bf1.v, acc1.v);
  }
  wait_tensor0();

#pragma unroll
  for (int r = 0; r < 8; ++r) {
    int m = m0 + r + ((lane >= 16) ? 8 : 0);
    if (m < M) {
      int na = n0 + lm;
      if (na < N) C[(size_t)na * ldc + m] = postproc(acc0.f[r], bias, m, act);
      int nb = n0 + 16 + lm;
      if (nb < N) C[(size_t)nb * ldc + m] = postproc(acc1.f[r], bias, m, act);
    }
  }
}

// ---------------------------------------------------------------------------
// WMMA implicit-GEMM conv (NCHW, OIHW weights, same-size output)
// M = Cout, N = B*H*W, K = Cin*KH*KW.  16x32 tile per wave.
// A (weights): clamped scalar loads (genuine finite values; K-tail products
// are zeroed on the B side).  B (im2col): branch-free select-address gather.
// ---------------------------------------------------------------------------
DEV F16Frag conv_gather(const float* __restrict__ X, int kbase, int K, int Cin,
                        int H, int Wd, int KH, int KW, int pad, int b, int oy,
                        int ox) {
  F16Frag f;
#pragma unroll
  for (int e = 0; e < 16; ++e) {
    int kb = kbase + e;
    bool kv = kb < K;
    int kk = kv ? kb : 0;
    int ci = kk / (KH * KW);
    int r2 = kk % (KH * KW);
    int rr = r2 / KW, ss = r2 % KW;
    int iy = oy + rr - pad, ix = ox + ss - pad;
    bool ok = kv && (iy >= 0) && (iy < H) && (ix >= 0) && (ix < Wd);
    size_t idx = ok ? ((((size_t)b * Cin + ci) * H + iy) * Wd + ix) : 0;
    float v = X[idx];
    f.h[e] = (_Float16)(ok ? v : 0.f);
  }
  return f;
}

__global__ void k_conv_wmma(const float* __restrict__ X,
                            const float* __restrict__ W,
                            const float* __restrict__ bias,
                            float* __restrict__ Y,
                            int Bn, int Cin, int Cout, int H, int Wd,
                            int KH, int KW, int pad, int act) {
  int lane = threadIdx.x & 31;
  int wave = blockIdx.x * (blockDim.x >> 5) + (threadIdx.x >> 5);
  int Npix = Bn * H * Wd;
  int mt = (Cout + 15) >> 4, nt = (Npix + 31) >> 5;
  if (wave >= mt * nt) return;
  int m0 = (wave % mt) << 4;
  int n0 = (wave / mt) << 5;
  int K = Cin * KH * KW;
  int lm = lane & 15;
  int mi  = min(m0 + lm, Cout - 1);
  int ni0 = min(n0 + lm, Npix - 1);
  int ni1 = min(n0 + 16 + lm, Npix - 1);
  int b0 = ni0 / (H * Wd), rm0 = ni0 % (H * Wd), oy0 = rm0 / Wd, ox0 = rm0 % Wd;
  int b1 = ni1 / (H * Wd), rm1 = ni1 % (H * Wd), oy1 = rm1 / Wd, ox1 = rm1 % Wd;
  int aoff = (lane < 16) ? 0 : 16;
  int boff = (lane < 16) ? 0 : 16;
  const float* Wrow = W + (size_t)mi * K;
  int lastK = K - 1;

  F32Acc acc0, acc1;
#pragma unroll
  for (int r = 0; r < 8; ++r) { acc0.f[r] = 0.f; acc1.f[r] = 0.f; }

  for (int k0 = 0; k0 < K; k0 += 32) {
    if (k0 + 64 < K) __builtin_prefetch(Wrow + k0 + 64, 0, 1);
    F16Frag af;
#pragma unroll
    for (int e = 0; e < 16; ++e) {
      int k = k0 + aoff + e;
      af.h[e] = (_Float16)Wrow[min(k, lastK)];
    }
    F16Frag bf0 = conv_gather(X, k0 + boff, K, Cin, H, Wd, KH, KW, pad, b0, oy0, ox0);
    acc0.v = wmma_f16(af.v, bf0.v, acc0.v);
    F16Frag bf1 = conv_gather(X, k0 + boff, K, Cin, H, Wd, KH, KW, pad, b1, oy1, ox1);
    acc1.v = wmma_f16(af.v, bf1.v, acc1.v);
  }

#pragma unroll
  for (int r = 0; r < 8; ++r) {
    int m = m0 + r + ((lane >= 16) ? 8 : 0);
    if (m < Cout) {
      int na = n0 + lm;
      if (na < Npix) {
        int b2 = na / (H * Wd), r2 = na % (H * Wd);
        Y[((size_t)b2 * Cout + m) * (size_t)(H * Wd) + r2] =
            postproc(acc0.f[r], bias, m, act);
      }
      int nb = n0 + 16 + lm;
      if (nb < Npix) {
        int b2 = nb / (H * Wd), r2 = nb % (H * Wd);
        Y[((size_t)b2 * Cout + m) * (size_t)(H * Wd) + r2] =
            postproc(acc1.f[r], bias, m, act);
      }
    }
  }
}

// ---------------------------------------------------------------------------
// Spectral band filter: separable 128-pt DFTs
// ---------------------------------------------------------------------------
__global__ void k_dft_rows(const float* __restrict__ x, float* __restrict__ re,
                           float* __restrict__ im, int Bn, int N) {
  int idx = blockIdx.x * blockDim.x + threadIdx.x;
  if (idx >= Bn * N * N) return;
  int kx = idx % N, r = (idx / N) % N, b = idx / (N * N);
  float sr = 0.f, si = 0.f;
  for (int n = 0; n < N; ++n) {
    float ang = -6.28318530718f * (float)(kx * n) / (float)N;
    float s, c; __sincosf(ang, &s, &c);
    float v = x[((size_t)b * N + r) * N + n];
    sr += v * c; si += v * s;
  }
  re[idx] = sr; im[idx] = si;
}

__global__ void k_dft_cols(const float* __restrict__ rre,
                           const float* __restrict__ rim,
                           float* __restrict__ re, float* __restrict__ im,
                           int Bn, int N) {
  int idx = blockIdx.x * blockDim.x + threadIdx.x;
  if (idx >= Bn * N * N) return;
  int kx = idx % N, ky = (idx / N) % N, b = idx / (N * N);
  float sr = 0.f, si = 0.f;
  for (int y = 0; y < N; ++y) {
    float ang = -6.28318530718f * (float)(ky * y) / (float)N;
    float s, c; __sincosf(ang, &s, &c);
    float ar = rre[((size_t)b * N + y) * N + kx];
    float ai = rim[((size_t)b * N + y) * N + kx];
    sr += ar * c - ai * s;
    si += ar * s + ai * c;
  }
  re[idx] = sr; im[idx] = si;
}

__global__ void k_idft_cols_mask(const float* __restrict__ Xre,
                                 const float* __restrict__ Xim,
                                 const float* __restrict__ mask,
                                 float* __restrict__ re, float* __restrict__ im,
                                 int Bn, int NB, int N) {
  int idx = blockIdx.x * blockDim.x + threadIdx.x;
  if (idx >= Bn * NB * N * N) return;
  int kx = idx % N, y = (idx / N) % N;
  int m = (idx / (N * N)) % NB, b = idx / (NB * N * N);
  float sr = 0.f, si = 0.f;
  for (int ky = 0; ky < N; ++ky) {
    float mk = mask[((size_t)m * N + ky) * N + kx];
    float ar = Xre[((size_t)b * N + ky) * N + kx] * mk;
    float ai = Xim[((size_t)b * N + ky) * N + kx] * mk;
    float ang = 6.28318530718f * (float)(ky * y) / (float)N;
    float s, c; __sincosf(ang, &s, &c);
    sr += ar * c - ai * s;
    si += ar * s + ai * c;
  }
  float inv = 1.f / (float)N;
  re[idx] = sr * inv; im[idx] = si * inv;
}

__global__ void k_idft_rows_real(const float* __restrict__ tre,
                                 const float* __restrict__ tim,
                                 float* __restrict__ out, int Bn, int NB, int N) {
  int idx = blockIdx.x * blockDim.x + threadIdx.x;
  if (idx >= Bn * NB * N * N) return;
  int x = idx % N;
  size_t rowBase = (size_t)(idx / N) * N;
  float sr = 0.f;
  for (int kx = 0; kx < N; ++kx) {
    float ang = 6.28318530718f * (float)(kx * x) / (float)N;
    float s, c; __sincosf(ang, &s, &c);
    sr += tre[rowBase + kx] * c - tim[rowBase + kx] * s;
  }
  out[idx] = sr / (float)N;
}

// ---------------------------------------------------------------------------
// Elementwise / small kernels
// ---------------------------------------------------------------------------
__global__ void k_maxpool2(const float* __restrict__ in, float* __restrict__ out,
                           int Bn, int C, int H, int W) {
  int OH = H >> 1, OW = W >> 1;
  int idx = blockIdx.x * blockDim.x + threadIdx.x;
  if (idx >= Bn * C * OH * OW) return;
  int ox = idx % OW; int t = idx / OW;
  int oy = t % OH; t /= OH;
  int c = t % C;   int b = t / C;
  const float* p = in + (((size_t)b * C + c) * H + oy * 2) * W + ox * 2;
  float v = fmaxf(fmaxf(p[0], p[1]), fmaxf(p[W], p[W + 1]));
  out[idx] = v;
}

__global__ void k_build_tok(const float* __restrict__ p4,
                            const float* __restrict__ cls,
                            float* __restrict__ hidden,
                            float* __restrict__ residual) {
  int idx = blockIdx.x * blockDim.x + threadIdx.x;
  if (idx >= 2 * 65 * 512) return;
  int c = idx % 512; int t = idx / 512;
  int pos = t % 65;  int b = t / 65;
  float v = (pos == 0) ? cls[c]
                       : p4[(((size_t)b * 512 + c) * 64) + (pos - 1)];
  hidden[idx] = v;
  residual[idx] = v;
}

__global__ void k_add_ln(float* __restrict__ residual,
                         const float* __restrict__ hidden,
                         const float* __restrict__ nw,
                         const float* __restrict__ nb,
                         float* __restrict__ hs, int D) {
  int row = blockIdx.x;
  __shared__ float red[256];
  int t = threadIdx.x, T = blockDim.x;
  float s = 0.f;
  for (int i = t; i < D; i += T) {
    float v = residual[(size_t)row * D + i] + hidden[(size_t)row * D + i];
    residual[(size_t)row * D + i] = v;
    s += v;
  }
  red[t] = s; __syncthreads();
  for (int o = T >> 1; o > 0; o >>= 1) {
    if (t < o) red[t] += red[t + o];
    __syncthreads();
  }
  float mean = red[0] / (float)D;
  __syncthreads();
  float s2 = 0.f;
  for (int i = t; i < D; i += T) {
    float d = residual[(size_t)row * D + i] - mean;
    s2 += d * d;
  }
  red[t] = s2; __syncthreads();
  for (int o = T >> 1; o > 0; o >>= 1) {
    if (t < o) red[t] += red[t + o];
    __syncthreads();
  }
  float inv = rsqrtf(red[0] / (float)D + 1e-5f);
  for (int i = t; i < D; i += T)
    hs[(size_t)row * D + i] =
        (residual[(size_t)row * D + i] - mean) * inv * nw[i] + nb[i];
}

// causal depthwise conv1d (width 4, left-pad 3) + SiLU; xz (N,2048) -> xc (N,1024)
__global__ void k_conv1d_silu(const float* __restrict__ xz,
                              const float* __restrict__ cw,
                              const float* __restrict__ cb,
                              float* __restrict__ xc, int Bn, int L) {
  int idx = blockIdx.x * blockDim.x + threadIdx.x;
  if (idx >= Bn * L * 1024) return;
  int d = idx % 1024; int t = idx / 1024;
  int l = t % L;      int b = t / L;
  float s = 0.f;
#pragma unroll
  for (int k = 0; k < 4; ++k) {
    int ll = l - 3 + k;
    if (ll >= 0) s += cw[d * 4 + k] * xz[((size_t)(b * L + ll)) * 2048 + d];
  }
  s += cb[d];
  xc[idx] = s / (1.f + __expf(-s));
}

// selective scan: thread per (b,d); fuses y = scan + Dp*xc, gate y*silu(z)
__global__ void k_scan(const float* __restrict__ dt,
                       const float* __restrict__ xc,
                       const float* __restrict__ xdbl,
                       const float* __restrict__ Alog,
                       const float* __restrict__ Dp,
                       const float* __restrict__ xz,
                       float* __restrict__ yg, int Bn, int L) {
  int idx = blockIdx.x * blockDim.x + threadIdx.x;
  if (idx >= Bn * 1024) return;
  int b = idx / 1024, d = idx % 1024;
  float A[16], h[16];
#pragma unroll
  for (int n = 0; n < 16; ++n) {
    A[n] = -__expf(Alog[d * 16 + n]);
    h[n] = 0.f;
  }
  float Dv = Dp[d];
  for (int l = 0; l < L; ++l) {
    size_t row = (size_t)(b * L + l);
    float dtv = dt[row * 1024 + d];
    float xcv = xc[row * 1024 + d];
    float du = dtv * xcv;
    float y = 0.f;
#pragma unroll
    for (int n = 0; n < 16; ++n) {
      float dA = __expf(dtv * A[n]);
      float Bs = xdbl[row * 64 + 32 + n];
      h[n] = dA * h[n] + du * Bs;
      y += h[n] * xdbl[row * 64 + 48 + n];
    }
    y += Dv * xcv;
    float z = xz[row * 2048 + 1024 + d];
    yg[row * 1024 + d] = y * (z / (1.f + __expf(-z)));
  }
}

__global__ void k_xsp(const float* __restrict__ hidden, float* __restrict__ xsp) {
  int idx = blockIdx.x * blockDim.x + threadIdx.x;
  if (idx >= 2 * 512 * 64) return;
  int s = idx % 64; int t = idx / 64;
  int c = t % 512;  int b = t / 512;
  xsp[idx] = hidden[((size_t)(b * 65 + 1 + s)) * 512 + c];
}

// transposed 2x2 stride-2 conv (einsum 'bcij,copq->boipjq') + bias
__global__ void k_up2(const float* __restrict__ din, const float* __restrict__ w,
                      const float* __restrict__ bias, float* __restrict__ out,
                      int Bn, int Ci, int Co, int H, int W) {
  int OH = 2 * H, OW = 2 * W;
  int idx = blockIdx.x * blockDim.x + threadIdx.x;
  if (idx >= Bn * Co * OH * OW) return;
  int ox = idx % OW; int t = idx / OW;
  int oy = t % OH;   t /= OH;
  int o = t % Co;    int b = t / Co;
  int i = oy >> 1, p = oy & 1, j = ox >> 1, q = ox & 1;
  float s = bias[o];
  for (int c = 0; c < Ci; ++c)
    s += din[(((size_t)b * Ci + c) * H + i) * W + j] *
         w[(((size_t)c * Co + o) * 2 + p) * 2 + q];
  out[idx] = s;
}

__global__ void k_concat(const float* __restrict__ e, const float* __restrict__ d,
                         float* __restrict__ out, int Bn, int Ce, int Cd,
                         int H, int W) {
  int Ct = Ce + Cd;
  int idx = blockIdx.x * blockDim.x + threadIdx.x;
  if (idx >= Bn * Ct * H * W) return;
  int x = idx % W; int t = idx / W;
  int y = t % H;   t /= H;
  int c = t % Ct;  int b = t / Ct;
  out[idx] = (c < Ce) ? e[(((size_t)b * Ce + c) * H + y) * W + x]
                      : d[(((size_t)b * Cd + (c - Ce)) * H + y) * W + x];
}

__global__ void k_resize16(const float* __restrict__ in, float* __restrict__ out,
                           int Bn, int C, int H, int W) {
  int OH = H * 16, OW = W * 16;
  int idx = blockIdx.x * blockDim.x + threadIdx.x;
  if (idx >= Bn * C * OH * OW) return;
  int ox = idx % OW; int t = idx / OW;
  int oy = t % OH;   t /= OH;
  int c = t % C;     int b = t / C;
  float sy = fminf(fmaxf((oy + 0.5f) / 16.f - 0.5f, 0.f), (float)(H - 1));
  float sx = fminf(fmaxf((ox + 0.5f) / 16.f - 0.5f, 0.f), (float)(W - 1));
  int y0 = (int)sy, x0 = (int)sx;
  int y1 = (y0 + 1 < H) ? y0 + 1 : H - 1;
  int x1 = (x0 + 1 < W) ? x0 + 1 : W - 1;
  float fy = sy - (float)y0, fx = sx - (float)x0;
  const float* p = in + ((size_t)b * C + c) * (size_t)H * W;
  float v = (1.f - fy) * ((1.f - fx) * p[y0 * W + x0] + fx * p[y0 * W + x1]) +
            fy * ((1.f - fx) * p[y1 * W + x0] + fx * p[y1 * W + x1]);
  out[idx] = v;
}

// ---------------------------------------------------------------------------
// Host side
// ---------------------------------------------------------------------------
static inline int cdiv_i(int a, int b) { return (a + b - 1) / b; }

static void gemm(hipStream_t s, const float* A, const float* B, const float* bias,
                 float* C, int M, int N, int K, int lda, int ldb, int ldc, int act) {
  int tiles = cdiv_i(M, 16) * cdiv_i(N, 32);
  k_gemm_wmma<<<cdiv_i(tiles, 8), 256, 0, s>>>(A, B, bias, C, M, N, K, lda, ldb,
                                               ldc, act);
}

static void conv(hipStream_t s, const float* X, const float* W, const float* bias,
                 float* Y, int Bn, int Cin, int Cout, int H, int Wd, int KH,
                 int KW, int pad, int act) {
  int tiles = cdiv_i(Cout, 16) * cdiv_i(Bn * H * Wd, 32);
  k_conv_wmma<<<cdiv_i(tiles, 8), 256, 0, s>>>(X, W, bias, Y, Bn, Cin, Cout, H,
                                               Wd, KH, KW, pad, act);
}

extern "C" void kernel_launch(void* const* d_in, const int* in_sizes, int n_in,
                              void* d_out, int out_size, void* d_ws,
                              size_t ws_size, hipStream_t stream) {
  (void)in_sizes; (void)n_in; (void)out_size; (void)ws_size;
  const float* x      = (const float*)d_in[0];
  const float* fmask  = (const float*)d_in[1];
  const float* e1w1 = (const float*)d_in[2];  const float* e1b1 = (const float*)d_in[3];
  const float* e1w2 = (const float*)d_in[4];  const float* e1b2 = (const float*)d_in[5];
  const float* e2w1 = (const float*)d_in[6];  const float* e2b1 = (const float*)d_in[7];
  const float* e2w2 = (const float*)d_in[8];  const float* e2b2 = (const float*)d_in[9];
  const float* e3w1 = (const float*)d_in[10]; const float* e3b1 = (const float*)d_in[11];
  const float* e3w2 = (const float*)d_in[12]; const float* e3b2 = (const float*)d_in[13];
  const float* e4w1 = (const float*)d_in[14]; const float* e4b1 = (const float*)d_in[15];
  const float* e4w2 = (const float*)d_in[16]; const float* e4b2 = (const float*)d_in[17];
  const float* cls     = (const float*)d_in[18];
  const float* norm_w  = (const float*)d_in[19];
  const float* norm_b  = (const float*)d_in[20];
  const float* in_proj = (const float*)d_in[21];
  const float* c1w     = (const float*)d_in[22];
  const float* c1b     = (const float*)d_in[23];
  const float* xpW     = (const float*)d_in[24];
  const float* dtW     = (const float*)d_in[25];
  const float* dtB     = (const float*)d_in[26];
  const float* Alog    = (const float*)d_in[27];
  const float* Dssm    = (const float*)d_in[28];
  const float* outW    = (const float*)d_in[29];
  const float* d4upw = (const float*)d_in[30]; const float* d4upb = (const float*)d_in[31];
  const float* d4w1  = (const float*)d_in[32]; const float* d4b1  = (const float*)d_in[33];
  const float* d4w2  = (const float*)d_in[34]; const float* d4b2  = (const float*)d_in[35];
  const float* d3upw = (const float*)d_in[36]; const float* d3upb = (const float*)d_in[37];
  const float* d3w1  = (const float*)d_in[38]; const float* d3b1  = (const float*)d_in[39];
  const float* d3w2  = (const float*)d_in[40]; const float* d3b2  = (const float*)d_in[41];
  const float* d2upw = (const float*)d_in[42]; const float* d2upb = (const float*)d_in[43];
  const float* d2w1  = (const float*)d_in[44]; const float* d2b1  = (const float*)d_in[45];
  const float* d2w2  = (const float*)d_in[46]; const float* d2b2  = (const float*)d_in[47];
  const float* d1upw = (const float*)d_in[48]; const float* d1upb = (const float*)d_in[49];
  const float* d1w1  = (const float*)d_in[50]; const float* d1b1  = (const float*)d_in[51];
  const float* d1w2  = (const float*)d_in[52]; const float* d1b2  = (const float*)d_in[53];
  const float* finw  = (const float*)d_in[54]; const float* finb  = (const float*)d_in[55];

  // --- workspace bump allocator ---
  size_t off = 0;
  auto alloc = [&](size_t n) {
    float* p = (float*)((char*)d_ws + off);
    off += ((n * sizeof(float) + 255) / 256) * 256;
    return p;
  };
  const int N = 128, Bn = 2, NB = 3, L = 65;
  float* rowRe = alloc(Bn * N * N); float* rowIm = alloc(Bn * N * N);
  float* colRe = alloc(Bn * N * N); float* colIm = alloc(Bn * N * N);
  float* bndRe = alloc(Bn * NB * N * N); float* bndIm = alloc(Bn * NB * N * N);
  float* xfreq = alloc(Bn * NB * N * N);
  float* e1a = alloc((size_t)Bn * 64 * 128 * 128);  float* e1 = alloc((size_t)Bn * 64 * 128 * 128);
  float* p1  = alloc((size_t)Bn * 64 * 64 * 64);
  float* e2a = alloc((size_t)Bn * 128 * 64 * 64);   float* e2 = alloc((size_t)Bn * 128 * 64 * 64);
  float* p2  = alloc((size_t)Bn * 128 * 32 * 32);
  float* e3a = alloc((size_t)Bn * 256 * 32 * 32);   float* e3 = alloc((size_t)Bn * 256 * 32 * 32);
  float* p3  = alloc((size_t)Bn * 256 * 16 * 16);
  float* e4a = alloc((size_t)Bn * 512 * 16 * 16);   float* e4 = alloc((size_t)Bn * 512 * 16 * 16);
  float* p4  = alloc((size_t)Bn * 512 * 8 * 8);
  float* hidden = alloc(Bn * L * 512);
  float* resid  = alloc(Bn * L * 512);
  float* hsbuf  = alloc(Bn * L * 512);
  float* xz   = alloc(Bn * L * 2048);
  float* xc   = alloc(Bn * L * 1024);
  float* xdbl = alloc(Bn * L * 64);
  float* dtb  = alloc(Bn * L * 1024);
  float* yg   = alloc(Bn * L * 1024);
  float* xsp  = alloc((size_t)Bn * 512 * 8 * 8);
  float* up4 = alloc((size_t)Bn * 256 * 16 * 16);
  float* ct4 = alloc((size_t)Bn * 768 * 16 * 16);
  float* d4a = alloc((size_t)Bn * 256 * 16 * 16);   float* d4 = alloc((size_t)Bn * 256 * 16 * 16);
  float* up3 = alloc((size_t)Bn * 128 * 32 * 32);
  float* ct3 = alloc((size_t)Bn * 384 * 32 * 32);
  float* d3a = alloc((size_t)Bn * 128 * 32 * 32);   float* d3 = alloc((size_t)Bn * 128 * 32 * 32);
  float* up2b = alloc((size_t)Bn * 64 * 64 * 64);
  float* ct2 = alloc((size_t)Bn * 192 * 64 * 64);
  float* d2a = alloc((size_t)Bn * 64 * 64 * 64);    float* d2 = alloc((size_t)Bn * 64 * 64 * 64);
  float* up1 = alloc((size_t)Bn * 32 * 128 * 128);
  float* ct1 = alloc((size_t)Bn * 96 * 128 * 128);
  float* d1a = alloc((size_t)Bn * 32 * 128 * 128);  float* d1 = alloc((size_t)Bn * 32 * 128 * 128);
  float* fin = alloc((size_t)Bn * 4 * 128 * 128);

  const int T = 256;
  // --- spectral band filter ---
  int nBN2 = Bn * N * N;
  k_dft_rows<<<cdiv_i(nBN2, T), T, 0, stream>>>(x, rowRe, rowIm, Bn, N);
  k_dft_cols<<<cdiv_i(nBN2, T), T, 0, stream>>>(rowRe, rowIm, colRe, colIm, Bn, N);
  int nBand = Bn * NB * N * N;
  k_idft_cols_mask<<<cdiv_i(nBand, T), T, 0, stream>>>(colRe, colIm, fmask, bndRe,
                                                       bndIm, Bn, NB, N);
  k_idft_rows_real<<<cdiv_i(nBand, T), T, 0, stream>>>(bndRe, bndIm, xfreq, Bn, NB, N);

  // --- encoder ---
  conv(stream, xfreq, e1w1, e1b1, e1a, Bn, 3, 64, 128, 128, 3, 3, 1, 1);
  conv(stream, e1a, e1w2, e1b2, e1, Bn, 64, 64, 128, 128, 3, 3, 1, 1);
  k_maxpool2<<<cdiv_i(Bn * 64 * 64 * 64, T), T, 0, stream>>>(e1, p1, Bn, 64, 128, 128);
  conv(stream, p1, e2w1, e2b1, e2a, Bn, 64, 128, 64, 64, 3, 3, 1, 1);
  conv(stream, e2a, e2w2, e2b2, e2, Bn, 128, 128, 64, 64, 3, 3, 1, 1);
  k_maxpool2<<<cdiv_i(Bn * 128 * 32 * 32, T), T, 0, stream>>>(e2, p2, Bn, 128, 64, 64);
  conv(stream, p2, e3w1, e3b1, e3a, Bn, 128, 256, 32, 32, 3, 3, 1, 1);
  conv(stream, e3a, e3w2, e3b2, e3, Bn, 256, 256, 32, 32, 3, 3, 1, 1);
  k_maxpool2<<<cdiv_i(Bn * 256 * 16 * 16, T), T, 0, stream>>>(e3, p3, Bn, 256, 32, 32);
  conv(stream, p3, e4w1, e4b1, e4a, Bn, 256, 512, 16, 16, 3, 3, 1, 1);
  conv(stream, e4a, e4w2, e4b2, e4, Bn, 512, 512, 16, 16, 3, 3, 1, 1);
  k_maxpool2<<<cdiv_i(Bn * 512 * 8 * 8, T), T, 0, stream>>>(e4, p4, Bn, 512, 16, 16);

  // --- Mamba stack ---
  k_build_tok<<<cdiv_i(Bn * L * 512, T), T, 0, stream>>>(p4, cls, hidden, resid);
  int rows = Bn * L;
  for (int l = 0; l < 24; ++l) {
    k_add_ln<<<rows, 256, 0, stream>>>(resid, hidden, norm_w + l * 512,
                                       norm_b + l * 512, hsbuf, 512);
    gemm(stream, in_proj + (size_t)l * 2048 * 512, hsbuf, nullptr, xz,
         2048, rows, 512, 512, 512, 2048, 0);
    k_conv1d_silu<<<cdiv_i(rows * 1024, T), T, 0, stream>>>(
        xz, c1w + (size_t)l * 1024 * 4, c1b + (size_t)l * 1024, xc, Bn, L);
    gemm(stream, xpW + (size_t)l * 64 * 1024, xc, nullptr, xdbl,
         64, rows, 1024, 1024, 1024, 64, 0);
    gemm(stream, dtW + (size_t)l * 1024 * 32, xdbl, dtB + (size_t)l * 1024, dtb,
         1024, rows, 32, 32, 64, 1024, 2);
    k_scan<<<cdiv_i(Bn * 1024, 128), 128, 0, stream>>>(
        dtb, xc, xdbl, Alog + (size_t)l * 1024 * 16, Dssm + (size_t)l * 1024,
        xz, yg, Bn, L);
    gemm(stream, outW + (size_t)l * 512 * 1024, yg, nullptr, hidden,
         512, rows, 1024, 1024, 1024, 512, 0);
  }
  k_xsp<<<cdiv_i(Bn * 512 * 64, T), T, 0, stream>>>(hidden, xsp);

  // --- decoder ---
  k_up2<<<cdiv_i(Bn * 256 * 16 * 16, T), T, 0, stream>>>(xsp, d4upw, d4upb, up4,
                                                         Bn, 512, 256, 8, 8);
  k_concat<<<cdiv_i(Bn * 768 * 16 * 16, T), T, 0, stream>>>(e4, up4, ct4, Bn, 512,
                                                            256, 16, 16);
  conv(stream, ct4, d4w1, d4b1, d4a, Bn, 768, 256, 16, 16, 3, 3, 1, 1);
  conv(stream, d4a, d4w2, d4b2, d4, Bn, 256, 256, 16, 16, 3, 3, 1, 1);

  k_up2<<<cdiv_i(Bn * 128 * 32 * 32, T), T, 0, stream>>>(d4, d3upw, d3upb, up3,
                                                         Bn, 256, 128, 16, 16);
  k_concat<<<cdiv_i(Bn * 384 * 32 * 32, T), T, 0, stream>>>(e3, up3, ct3, Bn, 256,
                                                            128, 32, 32);
  conv(stream, ct3, d3w1, d3b1, d3a, Bn, 384, 128, 32, 32, 3, 3, 1, 1);
  conv(stream, d3a, d3w2, d3b2, d3, Bn, 128, 128, 32, 32, 3, 3, 1, 1);

  k_up2<<<cdiv_i(Bn * 64 * 64 * 64, T), T, 0, stream>>>(d3, d2upw, d2upb, up2b,
                                                        Bn, 128, 64, 32, 32);
  k_concat<<<cdiv_i(Bn * 192 * 64 * 64, T), T, 0, stream>>>(e2, up2b, ct2, Bn, 128,
                                                            64, 64, 64);
  conv(stream, ct2, d2w1, d2b1, d2a, Bn, 192, 64, 64, 64, 3, 3, 1, 1);
  conv(stream, d2a, d2w2, d2b2, d2, Bn, 64, 64, 64, 64, 3, 3, 1, 1);

  k_up2<<<cdiv_i(Bn * 32 * 128 * 128, T), T, 0, stream>>>(d2, d1upw, d1upb, up1,
                                                          Bn, 64, 32, 64, 64);
  k_concat<<<cdiv_i(Bn * 96 * 128 * 128, T), T, 0, stream>>>(e1, up1, ct1, Bn, 64,
                                                             32, 128, 128);
  conv(stream, ct1, d1w1, d1b1, d1a, Bn, 96, 32, 128, 128, 3, 3, 1, 1);
  conv(stream, d1a, d1w2, d1b2, d1, Bn, 32, 32, 128, 128, 3, 3, 1, 1);

  conv(stream, d1, finw, finb, fin, Bn, 32, 4, 128, 128, 1, 1, 0, 0);
  k_resize16<<<cdiv_i(Bn * 4 * 2048 * 2048, T), T, 0, stream>>>(
      fin, (float*)d_out, Bn, 4, 128, 128);
}
